// DotProductAttention_17497696764367
// MI455X (gfx1250) — matrix operand order repaired and checked by reading, not compiled, not run on hardware
//
#include <hip/hip_runtime.h>
#include <math.h>

// ---------------- CDNA5 WMMA types ----------------
typedef __attribute__((ext_vector_type(16))) __bf16 v16bf;
typedef __attribute__((ext_vector_type(8)))  __bf16 v8bf;
typedef __attribute__((ext_vector_type(8)))  float  v8f;

#define WMMA_BF16(A, Bm, C) \
    __builtin_amdgcn_wmma_f32_16x16x32_bf16(false, (A), false, (Bm), (short)0, (C), false, false)

constexpr int BSZ = 8;
constexpr int TQ  = 2048;
constexpr int TV  = 2048;
constexpr int H   = 1024;
constexpr int H2  = 2048;

union AFrag { v16bf v; v8bf h[2]; };

// ---------------- Kernel 0a: fp32 -> bf16 elementwise ----------------
__global__ __launch_bounds__(256) void cvt_bf16(const float* __restrict__ src,
                                                __bf16* __restrict__ dst, int n) {
    int i = blockIdx.x * 256 + threadIdx.x;
    if (i < n) dst[i] = (__bf16)src[i];
}

// ---------------- Kernel 0b: V [B][TV][H] -> Vt bf16 [B][H][TV] (LDS tiled) ----
__global__ __launch_bounds__(256) void transpose_cvt(const float* __restrict__ V,
                                                     __bf16* __restrict__ Vt) {
    __shared__ __bf16 tile[32][33];
    const int v0 = blockIdx.x * 32, h0 = blockIdx.y * 32, b = blockIdx.z;
    const int tx = threadIdx.x;          // 0..31
    const int ty = threadIdx.y;          // 0..7
    #pragma unroll
    for (int i = ty; i < 32; i += 8)
        tile[i][tx] = (__bf16)V[((size_t)(b * TV + v0 + i)) * H + h0 + tx];
    __syncthreads();
    #pragma unroll
    for (int i = ty; i < 32; i += 8)
        Vt[((size_t)(b * H + h0 + i)) * TV + v0 + tx] = tile[tx][i];
}

// ---------------- Kernel 1: fused flash attention -> context (bf16) ----------
// 1 block (8 waves) = one (batch, 16-row q block). Online softmax with
// register stats, ping-pong LDS score tile, 2 barriers per KV iteration.
__global__ __launch_bounds__(256)
void attn_flash(const __bf16* __restrict__ Qb, const __bf16* __restrict__ Vb,
                const __bf16* __restrict__ Vtb, __bf16* __restrict__ ctxb) {
    __shared__ float S_lds[2][16][32];              // ping-pong partial-S tiles

    const int tid  = threadIdx.x;
    const int wave = tid >> 5;
    const int lane = tid & 31;
    const int half = lane >> 4;
    const int ln   = lane & 15;

    const int b  = blockIdx.x >> 7;                 // TQ/16 == 128
    const int qb = (blockIdx.x & 127) << 4;

    // ---- hoist Q A-fragments for this wave's 128-wide K slice (4 x k32) ----
    const int k0w = wave * 128;
    const __bf16* Qrow = Qb + ((size_t)(b * TQ + qb + ln)) * H;
    AFrag af[4];
    #pragma unroll
    for (int kk = 0; kk < 4; ++kk) {
        const __bf16* p = Qrow + k0w + kk * 32 + 8 * half;
        af[kk].h[0] = *(const v8bf*)(p);
        af[kk].h[1] = *(const v8bf*)(p + 16);
    }

    // zero both score buffers (1024 floats / 256 threads)
    #pragma unroll
    for (int z = 0; z < 4; ++z) ((float*)S_lds)[tid + 256 * z] = 0.f;
    __syncthreads();

    const int h0 = wave * 128;                      // this wave's slice of H
    v8f acc[8] = {};
    float M_run   = -3.0e38f;                       // running row max   (row ln)
    float Sum_run = 0.f;                            // running row denom (row ln)

    for (int kv0 = 0, p = 0; kv0 < TV; kv0 += 32, p ^= 1) {
        float (*Sb)[32] = S_lds[p];

        // ---- Phase A: partial S (16 x 32) over K slice [k0w, k0w+128) ----
        #pragma unroll
        for (int nt = 0; nt < 2; ++nt) {
            v8f s = {};
            const __bf16* vrow =
                Vb + ((size_t)(b * TV + kv0 + nt * 16 + ln)) * H + k0w + 8 * half;
            #pragma unroll
            for (int kk = 0; kk < 4; ++kk) {
                v16bf bfrag = __builtin_shufflevector(*(const v8bf*)(vrow + kk * 32), *(const v8bf*)(vrow + kk * 32 + 16), 0, 1, 2, 3, 4, 5, 6, 7, 8, 9, 10, 11, 12, 13, 14, 15);
                s = WMMA_BF16(af[kk].v, bfrag, s);
            }
            #pragma unroll
            for (int r = 0; r < 8; ++r)
                atomicAdd(&Sb[8 * half + r][nt * 16 + ln], s[r]);
        }
        // prefetch next KV tile (global_prefetch_b8; no counter cost)
        if (kv0 + 32 < TV) {
            __builtin_prefetch(
                Vb + ((size_t)(b * TV + kv0 + 32 + ln)) * H + k0w, 0, 0);
            __builtin_prefetch(
                Vtb + ((size_t)(b * H + h0 + ln)) * TV + kv0 + 32, 0, 0);
        }
        __syncthreads();

        // ---- Phase B+C (all waves, redundant wave-uniform stats) --------
        // This lane covers 16 of row ln's 32 columns (the pa pattern);
        // lane l^16 covers the other 16 -> one shfl_xor completes the row.
        const float* Srow = Sb[ln];
        float v0[8], v1[8];
        #pragma unroll
        for (int i = 0; i < 8; ++i) {
            v0[i] = Srow[8 * half + i];             // ds_load, static offsets
            v1[i] = Srow[16 + 8 * half + i];
        }
        float mx = -3.0e38f;
        #pragma unroll
        for (int i = 0; i < 8; ++i) { mx = fmaxf(mx, v0[i]); mx = fmaxf(mx, v1[i]); }
        mx = fmaxf(mx, __shfl_xor(mx, 16));
        float mnew = fmaxf(M_run, mx);
        float sc   = __expf(M_run - mnew);          // first iter: exp(-inf)=0
        float se   = 0.f;
        v16bf pa;                                   // P fragment (16x32 A-matrix)
        #pragma unroll
        for (int i = 0; i < 8; ++i) {
            float e0 = __expf(v0[i] - mnew);
            float e1 = __expf(v1[i] - mnew);
            se += e0 + e1;
            pa[i]     = (__bf16)e0;
            pa[i + 8] = (__bf16)e1;
        }
        se += __shfl_xor(se, 16);
        Sum_run = Sum_run * sc + se;
        M_run   = mnew;

        // per-row rescale: two constant-lane broadcasts + select on `half`
        float scr[8];
        #pragma unroll
        for (int r = 0; r < 8; ++r) {
            float s0 = __shfl(sc, r);
            float s1 = __shfl(sc, 8 + r);
            scr[r] = half ? s1 : s0;
        }
        #pragma unroll
        for (int t = 0; t < 8; ++t)
            #pragma unroll
            for (int r = 0; r < 8; ++r) acc[t][r] *= scr[r];

        #pragma unroll
        for (int t = 0; t < 8; ++t) {
            const __bf16* vt =
                Vtb + ((size_t)(b * H + h0 + t * 16 + ln)) * TV + kv0 + 8 * half;
            v16bf bfrag = __builtin_shufflevector(*(const v8bf*)vt, *(const v8bf*)(vt + 16), 0, 1, 2, 3, 4, 5, 6, 7, 8, 9, 10, 11, 12, 13, 14, 15);
            acc[t] = WMMA_BF16(pa, bfrag, acc[t]);
        }

        // zero the other buffer for iteration i+1 (overlaps with this phase)
        ((float*)S_lds[p ^ 1])[tid]       = 0.f;
        ((float*)S_lds[p ^ 1])[tid + 256] = 0.f;
        __syncthreads();
    }

    // ---- epilogue: normalize by softmax denom, store bf16 context ----
    float inv[8];
    #pragma unroll
    for (int r = 0; r < 8; ++r) {
        float s0 = __shfl(Sum_run, r);
        float s1 = __shfl(Sum_run, 8 + r);
        inv[r] = 1.0f / (half ? s1 : s0);
    }
    #pragma unroll
    for (int t = 0; t < 8; ++t)
        #pragma unroll
        for (int r = 0; r < 8; ++r) {
            int m = 8 * half + r;
            ctxb[((size_t)(b * TQ + qb + m)) * H + h0 + t * 16 + ln] =
                (__bf16)(acc[t][r] * inv[r]);
        }
}

// ---------------- Kernel 2: proj ([ctx|Q] @ W^T + bias) + LayerNorm ----------
__global__ __launch_bounds__(256)
void proj_ln(const __bf16* __restrict__ ctxb, const __bf16* __restrict__ Qb,
             const __bf16* __restrict__ Wb, const float* __restrict__ bias,
             const float* __restrict__ gamma, const float* __restrict__ beta,
             float* __restrict__ out) {
    __shared__ float ls_sum[16], ls_sq[16];
    const int tid  = threadIdx.x;
    const int wave = tid >> 5;
    const int lane = tid & 31;
    const int half = lane >> 4;
    const int ln   = lane & 15;
    const size_t rowbase = (size_t)blockIdx.x * 16;   // global row over B*TQ

    if (tid < 16) { ls_sum[tid] = 0.f; ls_sq[tid] = 0.f; }

    const int h0 = wave * 128;
    v8f acc[8] = {};
    const __bf16* Crow = ctxb + (rowbase + ln) * H;
    const __bf16* Qrow = Qb   + (rowbase + ln) * H;

    for (int k0 = 0; k0 < H2; k0 += 32) {
        const __bf16* src = (k0 < H) ? (Crow + k0) : (Qrow + (k0 - H));
        AFrag a;
        a.h[0] = *(const v8bf*)(src + 8 * half);
        a.h[1] = *(const v8bf*)(src + 8 * half + 16);
        #pragma unroll
        for (int t = 0; t < 8; ++t) {
            const __bf16* wp = Wb + (size_t)(h0 + t * 16 + ln) * H2 + k0 + 8 * half;
            v16bf bfrag = __builtin_shufflevector(*(const v8bf*)wp, *(const v8bf*)(wp + 16), 0, 1, 2, 3, 4, 5, 6, 7, 8, 9, 10, 11, 12, 13, 14, 15);
            acc[t] = WMMA_BF16(a.v, bfrag, acc[t]);
        }
    }

    // bias add
    #pragma unroll
    for (int t = 0; t < 8; ++t) {
        float bv = bias[h0 + t * 16 + ln];
        #pragma unroll
        for (int r = 0; r < 8; ++r) acc[t][r] += bv;
    }

    // LayerNorm reduction across the 1024 cols of each row
    __syncthreads();
    #pragma unroll
    for (int r = 0; r < 8; ++r) {
        float s = 0.f, q = 0.f;
        #pragma unroll
        for (int t = 0; t < 8; ++t) { float v = acc[t][r]; s += v; q += v * v; }
        atomicAdd(&ls_sum[8 * half + r], s);
        atomicAdd(&ls_sq [8 * half + r], q);
    }
    __syncthreads();

    #pragma unroll
    for (int r = 0; r < 8; ++r) {
        const int m = 8 * half + r;
        float mean = ls_sum[m] * (1.0f / H);
        float var  = ls_sq[m] * (1.0f / H) - mean * mean;
        float rstd = rsqrtf(var + 1e-5f);
        #pragma unroll
        for (int t = 0; t < 8; ++t) {
            int h = h0 + t * 16 + ln;
            out[(rowbase + m) * H + h] = (acc[t][r] - mean) * rstd * gamma[h] + beta[h];
        }
    }
}

// ---------------- host side ----------------
extern "C" void kernel_launch(void* const* d_in, const int* in_sizes, int n_in,
                              void* d_out, int out_size, void* d_ws, size_t ws_size,
                              hipStream_t stream) {
    const float* q     = (const float*)d_in[0];
    const float* v     = (const float*)d_in[1];
    const float* W     = (const float*)d_in[2];
    const float* bias  = (const float*)d_in[3];
    const float* gamma = (const float*)d_in[4];
    const float* beta  = (const float*)d_in[5];
    float* out = (float*)d_out;

    const size_t nq = (size_t)BSZ * TQ * H;   // 16.7M
    const size_t nv = (size_t)BSZ * TV * H;   // 16.7M
    const size_t nw = (size_t)H * H2;         // 2.1M

    __bf16* Qb = (__bf16*)d_ws;               // bf16 query
    __bf16* Vb = Qb + nq;                     // bf16 value (row-major [TV][H])
    __bf16* Vt = Vb + nv;                     // bf16 value transposed [H][TV]
    __bf16* Wb = Vt + nv;                     // bf16 W [H][2H]
    __bf16* Cb = Wb + nw;                     // bf16 context  (total ws ~138 MB)

    cvt_bf16<<<(int)((nq + 255) / 256), 256, 0, stream>>>(q, Qb, (int)nq);
    cvt_bf16<<<(int)((nv + 255) / 256), 256, 0, stream>>>(v, Vb, (int)nv);
    cvt_bf16<<<(int)((nw + 255) / 256), 256, 0, stream>>>(W, Wb, (int)nw);
    transpose_cvt<<<dim3(TV / 32, H / 32, BSZ), dim3(32, 8), 0, stream>>>(v, Vt);

    attn_flash<<<BSZ * (TQ / 16), 256, 0, stream>>>(Qb, Vb, Vt, Cb);
    proj_ln<<<(BSZ * TQ) / 16, 256, 0, stream>>>(Cb, Qb, Wb, bias, gamma, beta, out);
}
